// AnchorTs2Vec_4363686773048
// MI455X (gfx1250) — matrix operand order repaired and checked by reading, not compiled, not run on hardware
//
#include <hip/hip_runtime.h>

typedef __attribute__((ext_vector_type(16))) __bf16        bf16x16;
typedef __attribute__((ext_vector_type(8)))  float         f32x8;
typedef __attribute__((ext_vector_type(4)))  unsigned int  u32x4;
typedef __attribute__((ext_vector_type(8)))  unsigned int  u32x8;

// ---------- bf16 helpers (round-to-nearest-even split) ----------
static __device__ __forceinline__ unsigned short f32_to_bf16_rne(float x) {
    unsigned int u = __float_as_uint(x);
    u += 0x7FFFu + ((u >> 16) & 1u);
    return (unsigned short)(u >> 16);
}
static __device__ __forceinline__ float bf16_to_f32(unsigned short h) {
    return __uint_as_float(((unsigned int)h) << 16);
}

// ---------- operand loaders per CDNA5 WMMA VGPR layouts ----------
static __device__ __forceinline__ bf16x16 pack2(u32x4 a, u32x4 b) {
    u32x8 c;
    c[0] = a[0]; c[1] = a[1]; c[2] = a[2]; c[3] = a[3];
    c[4] = b[0]; c[5] = b[1]; c[6] = b[2]; c[7] = b[3];
    return __builtin_bit_cast(bf16x16, c);
}
// A 16x32 bf16: this lane's halves 0..7 = k0..k0+7, halves 8..15 = k0+16..k0+23
static __device__ __forceinline__ bf16x16 ldA(const unsigned short* p) {
    return pack2(*(const u32x4*)p, *(const u32x4*)(p + 16));
}
// B 32x16 bf16: this lane's halves 0..7 = k0..k0+7, halves 8..15 = k0+16..k0+23
static __device__ __forceinline__ bf16x16 ldB(const unsigned short* p) {
    return pack2(*(const u32x4*)p, *(const u32x4*)(p + 16));
}

static __device__ __forceinline__ f32x8 wmma_bf16(bf16x16 a, bf16x16 b, f32x8 c) {
    return __builtin_amdgcn_wmma_f32_16x16x32_bf16(
        /*neg_a=*/false, a, /*neg_b=*/false, b,
        /*c_mod=*/(short)0, c, /*reuse_a=*/false, /*reuse_b=*/false);
}

// ---------- kernel 1: split fp32 -> bf16 hi/lo ----------
__global__ void split_bf16x2(const float* __restrict__ X,
                             unsigned short* __restrict__ Hi,
                             unsigned short* __restrict__ Lo, int total) {
    int i = blockIdx.x * blockDim.x + threadIdx.x;
    if (i >= total) return;
    float x = X[i];
    unsigned short h = f32_to_bf16_rne(x);
    float rem = x - bf16_to_f32(h);
    Hi[i] = h;
    Lo[i] = f32_to_bf16_rne(rem);
}

// ---------- kernel 2: row squared norms (fp32) + init packed mins ----------
__global__ void row_sq_init(const float* __restrict__ X, float* __restrict__ sq,
                            unsigned long long* __restrict__ packed, int D) {
    __shared__ float s[256];
    int row = blockIdx.x, t = threadIdx.x;
    float x = X[(size_t)row * D + t];
    s[t] = x * x;
    __syncthreads();
    for (int off = 128; off > 0; off >>= 1) {
        if (t < off) s[t] += s[t + off];
        __syncthreads();
    }
    if (t == 0) {
        sq[row] = s[0];
        packed[row] = ~0ull;
    }
}

// ---------- kernel 3: WMMA Gram tiles + masked running argmin ----------
#define WAVES_PER_BLOCK 4
__global__ __launch_bounds__(32 * WAVES_PER_BLOCK)
void wmma_dist_argmin(const unsigned short* __restrict__ Ahi,
                      const unsigned short* __restrict__ Alo,
                      const float* __restrict__ sq,
                      const int* __restrict__ host,
                      unsigned long long* __restrict__ packed,
                      int N, int jTiles) {
    const int D = 256;       // K dimension (compile-time for full unroll)
    const int KB = D / 32;   // 8 k-blocks of 32

    const int lane = threadIdx.x & 31;
    const int wave = threadIdx.x >> 5;
    const int l15  = lane & 15;
    const int lH   = lane >> 4;

    const int i0 = (blockIdx.x * WAVES_PER_BLOCK + wave) * 16;  // row tile base
    const int jt0 = blockIdx.y * jTiles;                        // column chunk base

    // Hoist this wave's A tiles (hi+lo) into registers: reused for every j tile.
    const unsigned short* aRowH = Ahi + (size_t)(i0 + l15) * D;
    const unsigned short* aRowL = Alo + (size_t)(i0 + l15) * D;
    bf16x16 Ah[KB], Al[KB];
#pragma unroll
    for (int kb = 0; kb < KB; ++kb) {
        int k0 = kb * 32 + lH * 8;
        Ah[kb] = ldA(aRowH + k0);
        Al[kb] = ldA(aRowL + k0);
    }

    // Per-lane row metadata: this lane's C/D rows are m = r + 8*lH.
    const int giBase = i0 + lH * 8;
    float sqRow[8];
    int   hostRow[8];
#pragma unroll
    for (int r = 0; r < 8; ++r) {
        sqRow[r]   = sq[giBase + r];
        hostRow[r] = host[giBase + r];
    }

    unsigned long long best[8];
#pragma unroll
    for (int r = 0; r < 8; ++r) best[r] = ~0ull;

    for (int t = 0; t < jTiles; ++t) {
        const int j0 = (jt0 + t) * 16;
        const int gj = j0 + l15;  // this lane's column index (N dim)
        const unsigned short* bRowH = Ahi + (size_t)gj * D;
        const unsigned short* bRowL = Alo + (size_t)gj * D;

        f32x8 acc = {};  // fp32 accumulator shared by all three bf16 products
#pragma unroll
        for (int kb = 0; kb < KB; ++kb) {
            int k0 = kb * 32 + lH * 8;
            bf16x16 Bh = ldB(bRowH + k0);
            bf16x16 Bl = ldB(bRowL + k0);
            acc = wmma_bf16(Ah[kb], Bh, acc);  // hi*hi
            acc = wmma_bf16(Ah[kb], Bl, acc);  // hi*lo
            acc = wmma_bf16(Al[kb], Bh, acc);  // lo*hi
        }

        const float sqj   = sq[gj];
        const int   hostj = host[gj];
#pragma unroll
        for (int r = 0; r < 8; ++r) {
            const int gi = giBase + r;
            float d2 = fmaxf(sqRow[r] + sqj - 2.0f * acc[r], 0.0f);
            bool masked = (gi == gj) || (hostRow[r] == hostj);
            if (!masked) {
                unsigned long long key =
                    ((unsigned long long)__float_as_uint(d2) << 32) |
                    (unsigned int)gj;
                if (key < best[r]) best[r] = key;
            }
        }
    }

    // Reduce across the 16 lanes that share each row, then merge globally.
#pragma unroll
    for (int r = 0; r < 8; ++r) {
        unsigned long long v = best[r];
#pragma unroll
        for (int m = 8; m >= 1; m >>= 1) {
            unsigned long long o = __shfl_xor(v, m, 16);
            if (o < v) v = o;
        }
        if (l15 == 0) atomicMin(&packed[giBase + r], v);
    }
}

// ---------- kernel 4: gather e_an and emit (e_actv, e_ap, e_an) ----------
__global__ void finalize_out(const float* __restrict__ A, const float* __restrict__ P,
                             const unsigned long long* __restrict__ packed,
                             float* __restrict__ out, int N, int D) {
    const int row = blockIdx.x, t = threadIdx.x;  // D/4 float4 chunks per row
    unsigned int idx = (unsigned int)(packed[row] & 0xFFFFFFFFull);
    if (idx >= (unsigned int)N) idx = 0;
    const float4* a  = (const float4*)(A + (size_t)row * D);
    const float4* p  = (const float4*)(P + (size_t)row * D);
    const float4* an = (const float4*)(A + (size_t)idx * D);
    float4* o0 = (float4*)(out + (size_t)row * D);
    float4* o1 = (float4*)(out + (size_t)N * D + (size_t)row * D);
    float4* o2 = (float4*)(out + 2 * (size_t)N * D + (size_t)row * D);
    o0[t] = a[t];
    o1[t] = p[t];
    o2[t] = an[t];
}

extern "C" void kernel_launch(void* const* d_in, const int* in_sizes, int n_in,
                              void* d_out, int out_size, void* d_ws, size_t ws_size,
                              hipStream_t stream) {
    (void)n_in; (void)out_size; (void)ws_size;
    const float* e_actv = (const float*)d_in[0];
    const float* e_ap   = (const float*)d_in[1];
    const int*   host   = (const int*)d_in[2];
    float*       out    = (float*)d_out;

    const int N = in_sizes[2];            // 8192
    const int D = in_sizes[0] / N;        // 256 (kernels assume 256)
    const size_t nd = (size_t)N * D;

    // workspace layout
    unsigned short* Ahi = (unsigned short*)d_ws;   // nd bf16
    unsigned short* Alo = Ahi + nd;                // nd bf16
    float* sq = (float*)(Alo + nd);                // N floats
    unsigned long long* packed = (unsigned long long*)(sq + N);  // N u64

    // 1) split fp32 -> bf16 hi/lo
    {
        int total = (int)nd;
        int blocks = (total + 255) / 256;
        split_bf16x2<<<blocks, 256, 0, stream>>>(e_actv, Ahi, Alo, total);
    }
    // 2) fp32 row norms + init packed argmin keys
    row_sq_init<<<N, 256, 0, stream>>>(e_actv, sq, packed, D);

    // 3) WMMA Gram + masked argmin. 4 column chunks via grid.y, merged by atomicMin.
    {
        const int JCHUNKS = 4;
        int jTiles = (N / 16) / JCHUNKS;
        dim3 grid(N / (16 * WAVES_PER_BLOCK), JCHUNKS);
        wmma_dist_argmin<<<grid, 32 * WAVES_PER_BLOCK, 0, stream>>>(
            Ahi, Alo, sq, host, packed, N, jTiles);
    }

    // 4) gather negatives + write the 3 concatenated outputs
    finalize_out<<<N, D / 4, 0, stream>>>(e_actv, e_ap, packed, out, N, D);
}
